// SpatioTemporalBlock_10350871183669
// MI455X (gfx1250) — matrix operand order repaired and checked by reading, not compiled, not run on hardware
//
#include <hip/hip_runtime.h>

// ---------------------------------------------------------------------------
// Problem constants: x = (1, 256, 8, 48, 48) fp32
// ---------------------------------------------------------------------------
#define CCH   256
#define TT    8
#define HH    48
#define WW    48
#define HWSZ  (HH * WW)     // 2304
#define PSZ   (TT * HWSZ)   // 18432
#define OFFCH 72            // G*2*K
#define OFFPAD 80           // padded to M-tile granularity (16)
#define CG    64            // channels per deform group

typedef unsigned int   u32;
typedef unsigned short u16;
typedef __attribute__((ext_vector_type(16))) __bf16 bf16x16;
typedef __attribute__((ext_vector_type(8)))  float  f32x8;

// fragment-plane sizes (ushorts): [n_chunks * n_mtiles][32 lanes][16 elems]
#define TW_CHUNKS 216                      // 27 taps * 8 c-chunks
#define DW_CHUNKS 72                       // 36 (g,k) * 2 c-chunks
#define OW_CHUNKS 72                       // 9 taps * 8 c-chunks
#define TW_MT     16
#define DW_MT     16
#define OW_MT     5                        // ceil(72/16)
#define TW_ELEMS  (TW_CHUNKS * TW_MT * 512)
#define DW_ELEMS  (DW_CHUNKS * DW_MT * 512)
#define OW_ELEMS  (OW_CHUNKS * OW_MT * 512)

__device__ __forceinline__ int clampi(int v, int lo, int hi) {
    return v < lo ? lo : (v > hi ? hi : v);
}

// ---------------------------------------------------------------------------
// bf16 split helpers (bit-level)
// ---------------------------------------------------------------------------
__device__ __forceinline__ u16 f32_to_bf16(float f) {
    u32 u = __float_as_uint(f);
    return (u16)((u + 0x7FFFu + ((u >> 16) & 1u)) >> 16);   // RNE
}
__device__ __forceinline__ float bf16_to_f32(u16 h) {
    return __uint_as_float(((u32)h) << 16);
}
__device__ __forceinline__ void split2(float v, u16& hi, u16& lo) {
    hi = f32_to_bf16(v);
    lo = f32_to_bf16(v - bf16_to_f32(hi));
}
__device__ __forceinline__ u32 pack2(u16 a, u16 b) { return (u32)a | ((u32)b << 16); }

// ---------------------------------------------------------------------------
// Fragments in "fragment order": 32 lanes x 16 elems contiguous (1 KB block)
//   A 16x32: lane<16 -> row lane, K 0..15; lane+16 -> K 16..31
//   B 32x16: lane<16 -> K 0..15 (N=lane); lane+16 -> K 16..31
// ---------------------------------------------------------------------------
__device__ __forceinline__ bf16x16 load_frag_g(const u16* __restrict__ base,
                                               int fb, int lane) {
    union { bf16x16 v; uint4 q[2]; } u;
    const uint4* p = (const uint4*)(base + ((size_t)fb << 9) + (lane << 4));
    u.q[0] = p[0];
    u.q[1] = p[1];
    return u.v;
}
__device__ __forceinline__ bf16x16 load_frag_lds(const u16* Bs, int s, int lane) {
    union { bf16x16 v; uint4 q[2]; } u;
    const uint4* p = (const uint4*)(Bs + ((s * 32 + lane) << 4));
    u.q[0] = p[0];
    u.q[1] = p[1];
    return u.v;
}
__device__ __forceinline__ f32x8 wmma_bf16(bf16x16 a, bf16x16 b, f32x8 c) {
    return __builtin_amdgcn_wmma_f32_16x16x32_bf16(false, a, false, b,
                                                   (short)0, c, false, false);
}

// ---------------------------------------------------------------------------
// Conversion kernel 1: x (fp32) -> interleaved split plane xs = hi | lo<<16
// ---------------------------------------------------------------------------
__global__ __launch_bounds__(256)
void pack_x_kernel(const float* __restrict__ x, u32* __restrict__ xs) {
    const int i = blockIdx.x * 256 + threadIdx.x;
    u16 h, l;
    split2(x[i], h, l);
    xs[i] = pack2(h, l);
}

// ---------------------------------------------------------------------------
// Conversion kernels 2-4: weights -> hi/lo bf16 planes in fragment order.
//   m = mt*16 + (lane&15);  kb = (lane&16)?8:0;  kl = (kb<<1) + elem
// ---------------------------------------------------------------------------
__global__ __launch_bounds__(256)
void swz_tw_kernel(const float* __restrict__ w, u16* __restrict__ whi,
                   u16* __restrict__ wlo) {
    const int id   = blockIdx.x * 256 + threadIdx.x;
    const int fb   = id >> 9;
    const int lane = (id >> 4) & 31;
    const int elem = id & 15;
    const int cc   = fb >> 4;
    const int mt   = fb & 15;
    const int m    = mt * 16 + (lane & 15);
    const int kb   = (lane & 16) ? 8 : 0;
    const int kl   = (kb << 1) + elem;
    const int tap  = cc >> 3;
    const int c    = (cc & 7) * 32 + kl;
    u16 h, l;
    split2(w[(m * CCH + c) * 27 + tap], h, l);
    whi[id] = h; wlo[id] = l;
}

__global__ __launch_bounds__(256)
void swz_dw_kernel(const float* __restrict__ w, u16* __restrict__ whi,
                   u16* __restrict__ wlo) {
    const int id   = blockIdx.x * 256 + threadIdx.x;
    const int fb   = id >> 9;
    const int lane = (id >> 4) & 31;
    const int elem = id & 15;
    const int cc   = fb >> 4;
    const int mt   = fb & 15;
    const int m    = mt * 16 + (lane & 15);
    const int kb   = (lane & 16) ? 8 : 0;
    const int kl   = (kb << 1) + elem;
    const int gk   = cc >> 1;
    const int g    = gk / 9;
    const int k    = gk % 9;
    const int c    = g * CG + (cc & 1) * 32 + kl;
    u16 h, l;
    split2(w[(m * CCH + c) * 9 + k], h, l);
    whi[id] = h; wlo[id] = l;
}

__global__ __launch_bounds__(256)
void swz_ow_kernel(const float* __restrict__ w, u16* __restrict__ whi,
                   u16* __restrict__ wlo) {
    const int id   = blockIdx.x * 256 + threadIdx.x;
    const int fb   = id >> 9;
    const int lane = (id >> 4) & 31;
    const int elem = id & 15;
    const int cc   = fb / OW_MT;
    const int mt   = fb % OW_MT;
    const int m    = mt * 16 + (lane & 15);
    const int kb   = (lane & 16) ? 8 : 0;
    const int kl   = (kb << 1) + elem;
    const int tap  = cc >> 3;
    const int c    = (cc & 7) * 32 + kl;
    float v = (m < OFFCH) ? w[(m * CCH + c) * 9 + tap] : 0.0f;
    u16 h, l;
    split2(v, h, l);
    whi[id] = h; wlo[id] = l;
}

// ---------------------------------------------------------------------------
// Kernel A: offset conv (1x3x3, pad (0,1,1)).  Tile 16(M) x 128(N), 8 waves.
// Offset plane is padded to 80 channels -> unconditional epilogue stores.
// ---------------------------------------------------------------------------
__global__ __launch_bounds__(256)
void offset_conv_kernel(const u32* __restrict__ xs,
                        const u16* __restrict__ owhi, const u16* __restrict__ owlo,
                        float* __restrict__ off_ws) {    // [80][PSZ]
    __shared__ alignas(16) u16 Bs_hi[8 * 32 * 16];
    __shared__ alignas(16) u16 Bs_lo[8 * 32 * 16];

    const int tid  = threadIdx.x;
    const int lane = tid & 31;
    const int wid  = tid >> 5;
    const int m0   = blockIdx.y * 16;
    const int p0   = blockIdx.x * 128;

    const int j     = tid & 127;
    const int p     = p0 + j;
    const int pt    = p / HWSZ;
    const int py    = (p % HWSZ) / WW;
    const int px    = p % WW;
    const int rhalf = (tid >> 7) * 16;
    const int sb    = j >> 4;
    const int lt    = (j & 15) + rhalf;

    f32x8 acc = {};

    int tap = 0;
    for (int dy = -1; dy <= 1; ++dy)
    for (int dx = -1; dx <= 1; ++dx, ++tap) {
        const int  yy    = py + dy, xx = px + dx;
        const bool valid = ((unsigned)yy < HH) && ((unsigned)xx < WW);
        const int  ofs   = pt * HWSZ + clampi(yy, 0, HH - 1) * WW
                                     + clampi(xx, 0, WW - 1);
        for (int cc = 0; cc < 8; ++cc) {
            const int cbase = cc * 32;
            __syncthreads();
#pragma unroll
            for (int g8 = 0; g8 < 2; ++g8) {
                u32 hh[4], ll[4];
#pragma unroll
                for (int i = 0; i < 4; ++i) {
                    const int r = rhalf + g8 * 8 + 2 * i;
                    u32 v0 = xs[(cbase + r)     * PSZ + ofs];
                    u32 v1 = xs[(cbase + r + 1) * PSZ + ofs];
                    v0 = valid ? v0 : 0u;
                    v1 = valid ? v1 : 0u;
                    hh[i] = (v0 & 0xFFFFu) | (v1 << 16);
                    ll[i] = (v0 >> 16)     | (v1 & 0xFFFF0000u);
                }
                uint4 qh; qh.x = hh[0]; qh.y = hh[1]; qh.z = hh[2]; qh.w = hh[3];
                uint4 ql; ql.x = ll[0]; ql.y = ll[1]; ql.z = ll[2]; ql.w = ll[3];
                *(uint4*)(Bs_hi + ((sb * 32 + lt) << 4) + g8 * 8) = qh;
                *(uint4*)(Bs_lo + ((sb * 32 + lt) << 4) + g8 * 8) = ql;
            }
            __syncthreads();

            const int fb = (tap * 8 + cc) * OW_MT + blockIdx.y;
            bf16x16 ah = load_frag_g(owhi, fb, lane);
            bf16x16 al = load_frag_g(owlo, fb, lane);
            bf16x16 bh = load_frag_lds(Bs_hi, wid, lane);
            bf16x16 bl = load_frag_lds(Bs_lo, wid, lane);
            acc = wmma_bf16(ah, bh, acc);
            acc = wmma_bf16(ah, bl, acc);
            acc = wmma_bf16(al, bh, acc);
        }
    }

    const int col = lane & 15;
    const int rb  = (lane >> 4) << 3;
#pragma unroll
    for (int i = 0; i < 8; ++i) {
        const int m = m0 + rb + i;                 // < 80, plane is padded
        off_ws[m * PSZ + p0 + wid * 16 + col] = acc[i];
    }
}

// ---------------------------------------------------------------------------
// Kernel B: deformable conv (one frame per blockIdx.z). Tile 64(M) x 64(N).
// One stage per (g,k): K=64, bilinear params once, branchless masked gather.
// ---------------------------------------------------------------------------
__global__ __launch_bounds__(256)
void deform_conv_kernel(const float* __restrict__ x,
                        const float* __restrict__ off,     // [80][PSZ], 0..71 used
                        const u16* __restrict__ dwhi, const u16* __restrict__ dwlo,
                        float* __restrict__ out) {
    __shared__ alignas(16) u16 Bs_hi[8 * 32 * 16];
    __shared__ alignas(16) u16 Bs_lo[8 * 32 * 16];

    const int tid   = threadIdx.x;
    const int lane  = tid & 31;
    const int wid   = tid >> 5;
    const int m_sub = wid & 3;
    const int n0    = (wid >> 2) * 2;

    const int t  = blockIdx.z;
    const int m0 = blockIdx.y * 64;
    const int p0 = blockIdx.x * 64;

    const int j     = tid & 63;
    const int p     = p0 + j;
    const int cy    = p / WW;
    const int cx    = p % WW;
    const int rbase = (tid >> 6) * 8;
    const int sb    = j >> 4;
    const int lt    = (j & 15) + (rbase & 16);
    const int eb    = rbase & 8;

    f32x8 acc0 = {}, acc1 = {};

    int gk = 0;
    for (int g = 0; g < 4; ++g)
    for (int ky = 0; ky < 3; ++ky)
    for (int kx = 0; kx < 3; ++kx, ++gk) {
        const float dyv = off[(gk * 2)     * PSZ + t * HWSZ + p];
        const float dxv = off[(gk * 2 + 1) * PSZ + t * HWSZ + p];
        const float fpy = (float)(cy - 1 + ky) + dyv;
        const float fpx = (float)(cx - 1 + kx) + dxv;
        const float fy  = floorf(fpy), fx = floorf(fpx);
        const int   y0  = (int)fy,     x0 = (int)fx;
        const float wy  = fpy - fy,    wx = fpx - fx;
        const bool  vy0 = (unsigned)y0       < HH;
        const bool  vy1 = (unsigned)(y0 + 1) < HH;
        const bool  vx0 = (unsigned)x0       < WW;
        const bool  vx1 = (unsigned)(x0 + 1) < WW;
        const float m00 = (vy0 && vx0) ? (1.f - wy) * (1.f - wx) : 0.f;
        const float m01 = (vy0 && vx1) ? (1.f - wy) * wx         : 0.f;
        const float m10 = (vy1 && vx0) ? wy * (1.f - wx)         : 0.f;
        const float m11 = (vy1 && vx1) ? wy * wx                 : 0.f;
        const int iy0 = clampi(y0, 0, HH - 1), iy1 = clampi(y0 + 1, 0, HH - 1);
        const int ix0 = clampi(x0, 0, WW - 1), ix1 = clampi(x0 + 1, 0, WW - 1);
        const int o00 = iy0 * WW + ix0, o01 = iy0 * WW + ix1;
        const int o10 = iy1 * WW + ix0, o11 = iy1 * WW + ix1;

        __syncthreads();
#pragma unroll
        for (int q = 0; q < 2; ++q) {                  // two 32-row subchunks
            const int cb = g * CG + q * 32;
            u32 hh[4], ll[4];
#pragma unroll
            for (int i = 0; i < 4; ++i) {
                u16 hv[2], lv[2];
#pragma unroll
                for (int qq = 0; qq < 2; ++qq) {
                    const int r = rbase + 2 * i + qq;
                    const float* bp = x + (cb + r) * PSZ + t * HWSZ;
                    const float v = m00 * bp[o00] + m01 * bp[o01]
                                  + m10 * bp[o10] + m11 * bp[o11];
                    split2(v, hv[qq], lv[qq]);
                }
                hh[i] = pack2(hv[0], hv[1]);
                ll[i] = pack2(lv[0], lv[1]);
            }
            uint4 qh; qh.x = hh[0]; qh.y = hh[1]; qh.z = hh[2]; qh.w = hh[3];
            uint4 ql; ql.x = ll[0]; ql.y = ll[1]; ql.z = ll[2]; ql.w = ll[3];
            *(uint4*)(Bs_hi + (((q * 4 + sb) * 32 + lt) << 4) + eb) = qh;
            *(uint4*)(Bs_lo + (((q * 4 + sb) * 32 + lt) << 4) + eb) = ql;
        }
        __syncthreads();

#pragma unroll
        for (int q = 0; q < 2; ++q) {
            const int fb = (gk * 2 + q) * DW_MT + blockIdx.y * 4 + m_sub;
            bf16x16 ah  = load_frag_g(dwhi, fb, lane);
            bf16x16 al  = load_frag_g(dwlo, fb, lane);
            bf16x16 b0h = load_frag_lds(Bs_hi, q * 4 + n0, lane);
            bf16x16 b0l = load_frag_lds(Bs_lo, q * 4 + n0, lane);
            bf16x16 b1h = load_frag_lds(Bs_hi, q * 4 + n0 + 1, lane);
            bf16x16 b1l = load_frag_lds(Bs_lo, q * 4 + n0 + 1, lane);
            acc0 = wmma_bf16(ah, b0h, acc0);
            acc0 = wmma_bf16(ah, b0l, acc0);
            acc0 = wmma_bf16(al, b0h, acc0);
            acc1 = wmma_bf16(ah, b1h, acc1);
            acc1 = wmma_bf16(ah, b1l, acc1);
            acc1 = wmma_bf16(al, b1h, acc1);
        }
    }

    const int col = lane & 15;
    const int rb  = (lane >> 4) << 3;
#pragma unroll
    for (int i = 0; i < 8; ++i) {
        const int m = m0 + m_sub * 16 + rb + i;
        out[m * PSZ + t * HWSZ + p0 + n0 * 16 + col]       = acc0[i];
        out[m * PSZ + t * HWSZ + p0 + (n0 + 1) * 16 + col] = acc1[i];
    }
}

// ---------------------------------------------------------------------------
// Kernel C: temporal 3x3x3 conv; out += acc + bias.  Tile 64x64, K=64 stages.
// ---------------------------------------------------------------------------
__global__ __launch_bounds__(256)
void temporal_conv_kernel(const u32* __restrict__ xs,
                          const u16* __restrict__ twhi, const u16* __restrict__ twlo,
                          const float* __restrict__ bias,
                          float* __restrict__ out) {
    __shared__ alignas(16) u16 Bs_hi[8 * 32 * 16];
    __shared__ alignas(16) u16 Bs_lo[8 * 32 * 16];

    const int tid   = threadIdx.x;
    const int lane  = tid & 31;
    const int wid   = tid >> 5;
    const int m_sub = wid & 3;
    const int n0    = (wid >> 2) * 2;

    const int m0 = blockIdx.y * 64;
    const int p0 = blockIdx.x * 64;

    const int j     = tid & 63;
    const int p     = p0 + j;
    const int pt    = p / HWSZ;
    const int py    = (p % HWSZ) / WW;
    const int px    = p % WW;
    const int rbase = (tid >> 6) * 8;
    const int sb    = j >> 4;
    const int lt    = (j & 15) + (rbase & 16);
    const int eb    = rbase & 8;

    f32x8 acc0 = {}, acc1 = {};

    int tap = 0;
    for (int dt = -1; dt <= 1; ++dt)
    for (int dy = -1; dy <= 1; ++dy)
    for (int dx = -1; dx <= 1; ++dx, ++tap) {
        const int  tc    = pt + dt, yy = py + dy, xx = px + dx;
        const bool valid = ((unsigned)tc < TT) && ((unsigned)yy < HH) &&
                           ((unsigned)xx < WW);
        const int  ofs   = clampi(tc, 0, TT - 1) * HWSZ
                         + clampi(yy, 0, HH - 1) * WW
                         + clampi(xx, 0, WW - 1);
        for (int pr = 0; pr < 4; ++pr) {               // K=64 stage
            const int cc0 = pr * 2;

            const int nfb = (tap * 8 + cc0 + 2) * TW_MT + blockIdx.y * 4 + m_sub;
            if (nfb < TW_CHUNKS * TW_MT)
                __builtin_prefetch(twhi + ((size_t)nfb << 9), 0, 1);

            __syncthreads();
#pragma unroll
            for (int q = 0; q < 2; ++q) {
                const int cbase = (cc0 + q) * 32;
                u32 hh[4], ll[4];
#pragma unroll
                for (int i = 0; i < 4; ++i) {
                    const int r = rbase + 2 * i;
                    u32 v0 = xs[(cbase + r)     * PSZ + ofs];
                    u32 v1 = xs[(cbase + r + 1) * PSZ + ofs];
                    v0 = valid ? v0 : 0u;
                    v1 = valid ? v1 : 0u;
                    hh[i] = (v0 & 0xFFFFu) | (v1 << 16);
                    ll[i] = (v0 >> 16)     | (v1 & 0xFFFF0000u);
                }
                uint4 qh; qh.x = hh[0]; qh.y = hh[1]; qh.z = hh[2]; qh.w = hh[3];
                uint4 ql; ql.x = ll[0]; ql.y = ll[1]; ql.z = ll[2]; ql.w = ll[3];
                *(uint4*)(Bs_hi + (((q * 4 + sb) * 32 + lt) << 4) + eb) = qh;
                *(uint4*)(Bs_lo + (((q * 4 + sb) * 32 + lt) << 4) + eb) = ql;
            }
            __syncthreads();

#pragma unroll
            for (int q = 0; q < 2; ++q) {
                const int fb = (tap * 8 + cc0 + q) * TW_MT + blockIdx.y * 4 + m_sub;
                bf16x16 ah  = load_frag_g(twhi, fb, lane);
                bf16x16 al  = load_frag_g(twlo, fb, lane);
                bf16x16 b0h = load_frag_lds(Bs_hi, q * 4 + n0, lane);
                bf16x16 b0l = load_frag_lds(Bs_lo, q * 4 + n0, lane);
                bf16x16 b1h = load_frag_lds(Bs_hi, q * 4 + n0 + 1, lane);
                bf16x16 b1l = load_frag_lds(Bs_lo, q * 4 + n0 + 1, lane);
                acc0 = wmma_bf16(ah, b0h, acc0);
                acc0 = wmma_bf16(ah, b0l, acc0);
                acc0 = wmma_bf16(al, b0h, acc0);
                acc1 = wmma_bf16(ah, b1h, acc1);
                acc1 = wmma_bf16(ah, b1l, acc1);
                acc1 = wmma_bf16(al, b1h, acc1);
            }
        }
    }

    const int col = lane & 15;
    const int rb  = (lane >> 4) << 3;
#pragma unroll
    for (int i = 0; i < 8; ++i) {
        const int m  = m0 + m_sub * 16 + rb + i;
        const float b = bias[m];
        out[m * PSZ + p0 + n0 * 16 + col]       += acc0[i] + b;
        out[m * PSZ + p0 + (n0 + 1) * 16 + col] += acc1[i] + b;
    }
}

// ---------------------------------------------------------------------------
// Launch: pack/swizzle -> offsets -> deform -> temporal
// ---------------------------------------------------------------------------
extern "C" void kernel_launch(void* const* d_in, const int* in_sizes, int n_in,
                              void* d_out, int out_size, void* d_ws, size_t ws_size,
                              hipStream_t stream) {
    const float* x  = (const float*)d_in[0];   // (1,256,8,48,48)
    const float* ow = (const float*)d_in[1];   // (72,256,1,3,3)
    const float* dw = (const float*)d_in[2];   // (256,256,3,3)
    const float* tw = (const float*)d_in[3];   // (256,256,3,3,3)
    const float* tb = (const float*)d_in[4];   // (256,)
    float* out = (float*)d_out;

    // workspace partition (~35 MB total)
    char* wp = (char*)d_ws;
    float* off = (float*)wp;  wp += (size_t)OFFPAD * PSZ * sizeof(float);
    u32*  xs   = (u32*)wp;    wp += (size_t)CCH * PSZ * sizeof(u32);
    u16*  twhi = (u16*)wp;    wp += (size_t)TW_ELEMS * sizeof(u16);
    u16*  twlo = (u16*)wp;    wp += (size_t)TW_ELEMS * sizeof(u16);
    u16*  dwhi = (u16*)wp;    wp += (size_t)DW_ELEMS * sizeof(u16);
    u16*  dwlo = (u16*)wp;    wp += (size_t)DW_ELEMS * sizeof(u16);
    u16*  owhi = (u16*)wp;    wp += (size_t)OW_ELEMS * sizeof(u16);
    u16*  owlo = (u16*)wp;

    dim3 blk(256, 1, 1);

    pack_x_kernel<<<dim3(CCH * PSZ / 256), blk, 0, stream>>>(x, xs);
    swz_tw_kernel<<<dim3(TW_ELEMS / 256), blk, 0, stream>>>(tw, twhi, twlo);
    swz_dw_kernel<<<dim3(DW_ELEMS / 256), blk, 0, stream>>>(dw, dwhi, dwlo);
    swz_ow_kernel<<<dim3(OW_ELEMS / 256), blk, 0, stream>>>(ow, owhi, owlo);

    offset_conv_kernel<<<dim3(PSZ / 128, OW_MT), blk, 0, stream>>>(xs, owhi, owlo, off);
    deform_conv_kernel<<<dim3(HWSZ / 64, CCH / 64, TT), blk, 0, stream>>>(
        x, off, dwhi, dwlo, out);
    temporal_conv_kernel<<<dim3(PSZ / 64, CCH / 64), blk, 0, stream>>>(
        xs, twhi, twlo, tb, out);
}
